// Model_87960930222090
// MI455X (gfx1250) — matrix operand order repaired and checked by reading, not compiled, not run on hardware
//
#include <hip/hip_runtime.h>
#include <hip/hip_bf16.h>

#define Bn   128
#define Ln   512
#define Dn   128
#define DFFn 512
#define ELn  2
#define TOPK 18

typedef __attribute__((ext_vector_type(8)))  _Float16 h8;
typedef __attribute__((ext_vector_type(16))) _Float16 v16h;
typedef __attribute__((ext_vector_type(8)))  float    v8f;
typedef int v4i __attribute__((vector_size(16)));

#if defined(__has_builtin)
#  if __has_builtin(__builtin_amdgcn_global_load_async_to_lds_b128)
#    define HAVE_ASYNC_LDS 1
#  endif
#endif
#ifndef HAVE_ASYNC_LDS
#  define HAVE_ASYNC_LDS 0
#endif

#if HAVE_ASYNC_LDS
#define GLOBAL_V4I(p) ((__attribute__((address_space(1))) v4i*)(p))
#define LDS_V4I(p)    ((__attribute__((address_space(3))) v4i*)(p))
__device__ __forceinline__ void wait_async_lds() {
#  if __has_builtin(__builtin_amdgcn_s_wait_asynccnt)
  __builtin_amdgcn_s_wait_asynccnt(0);
#  else
  asm volatile("s_wait_asynccnt 0" ::: "memory");
#  endif
}
#endif

__device__ __forceinline__ float gelu_exact(float x) {
  return 0.5f * x * (1.0f + erff(x * 0.70710678118654752f));
}

__device__ __forceinline__ v16h frag16(const _Float16* p0, const _Float16* p1) {
  h8 lo = *(const h8*)p0;
  h8 hi = *(const h8*)p1;
  return __builtin_shufflevector(lo, hi,
      0,1,2,3,4,5,6,7,8,9,10,11,12,13,14,15);
}

// ---------------------------------------------------------------------------
// Generic WMMA GEMM:  C[m,n] = act( sum_k A[rowmap(m),k] * W[n,k] + bias[n] (+C) )
// 128x128 block tile, 8 waves x (32M x 64N) = 8 WMMA per wave per K-step,
// double-buffered LDS; staging via GLOBAL_LOAD_ASYNC_TO_LDS_B128 when available.
// A rows grouped in batches of Lrows with circular shift (circular conv).
// Per-z batching via strideA/strideW/strideC. M,N mult of 128; K mult of 32.
// ---------------------------------------------------------------------------
__global__ __launch_bounds__(256)
void gemm_wmma(const _Float16* __restrict__ A, int lda, long long strideA,
               const _Float16* __restrict__ W, int ldw, long long strideW,
               const float* __restrict__ bias,
               float* __restrict__ Cf, _Float16* __restrict__ Ch, long long strideC,
               int M, int N, int K, int Lrows, int shift, int accFlag, int actFlag)
{
  __shared__ _Float16 As[2][128][40];
  __shared__ _Float16 Ws[2][128][40];

  const int tid  = threadIdx.x;
  const int lane = tid & 31;
  const int wv   = tid >> 5;          // wave 0..7
  const int wm   = (wv & 3) * 32;     // M sub-tile of wave (2 x 16)
  const int wn   = (wv >> 2) * 64;    // N sub-tile of wave (4 x 16)
  const int z    = blockIdx.z;
  const int m0   = blockIdx.y * 128;
  const int n0   = blockIdx.x * 128;

  // cooperative staging: row = tid/2, 16 halves at col (tid&1)*16
  const int ra = tid >> 1;
  const int cg = (tid & 1) * 16;
  int mG = m0 + ra;
  int bb = mG / Lrows;
  int ll = mG - bb * Lrows + shift;
  if (ll < 0)       ll += Lrows;
  if (ll >= Lrows)  ll -= Lrows;
  const _Float16* Arow = A + (size_t)z * strideA + ((size_t)bb * Lrows + ll) * lda;
  const _Float16* Wrow = W + (size_t)z * strideW + (size_t)(n0 + ra) * ldw;

  v8f acc[2][4] = {};

  auto stage = [&](int buf, int k0) {
#if HAVE_ASYNC_LDS
    __builtin_amdgcn_global_load_async_to_lds_b128(
        GLOBAL_V4I(Arow + k0 + cg),     LDS_V4I(&As[buf][ra][cg]),     0, 0);
    __builtin_amdgcn_global_load_async_to_lds_b128(
        GLOBAL_V4I(Arow + k0 + cg + 8), LDS_V4I(&As[buf][ra][cg + 8]), 0, 0);
    __builtin_amdgcn_global_load_async_to_lds_b128(
        GLOBAL_V4I(Wrow + k0 + cg),     LDS_V4I(&Ws[buf][ra][cg]),     0, 0);
    __builtin_amdgcn_global_load_async_to_lds_b128(
        GLOBAL_V4I(Wrow + k0 + cg + 8), LDS_V4I(&Ws[buf][ra][cg + 8]), 0, 0);
#else
    h8 a0 = *(const h8*)(Arow + k0 + cg);
    h8 a1 = *(const h8*)(Arow + k0 + cg + 8);
    h8 w0 = *(const h8*)(Wrow + k0 + cg);
    h8 w1 = *(const h8*)(Wrow + k0 + cg + 8);
    *(h8*)&As[buf][ra][cg]     = a0;
    *(h8*)&As[buf][ra][cg + 8] = a1;
    *(h8*)&Ws[buf][ra][cg]     = w0;
    *(h8*)&Ws[buf][ra][cg + 8] = w1;
#endif
  };

  // A fragment: lanes 0-15 M rows, K {koff..koff+7, 16+koff..}; lanes 16-31 koff=8
  const int am   = wm + (lane & 15);
  const int koff = (lane >> 4) * 8;
  // B fragment: B[k][n]=W[n][k]; same K order as A (lanes 16-31 kb=8)
  const int kb   = (lane >> 4) * 8;
  const int nnb  = wn + (lane & 15);

  stage(0, 0);
  int buf = 0;
  for (int k0 = 0; k0 < K; k0 += 32) {
#if HAVE_ASYNC_LDS
    wait_async_lds();               // own wave's async copies into LDS done
#endif
    __syncthreads();                // all waves' staging visible

    if (k0 + 32 < K) stage(buf ^ 1, k0 + 32);   // overlap next stage with math

    v16h a0 = frag16(&As[buf][am][koff],       &As[buf][am][16 + koff]);
    v16h a1 = frag16(&As[buf][am + 16][koff],  &As[buf][am + 16][16 + koff]);
    v16h b0 = frag16(&Ws[buf][nnb][kb],        &Ws[buf][nnb][16 + kb]);
    v16h b1 = frag16(&Ws[buf][nnb + 16][kb],   &Ws[buf][nnb + 16][16 + kb]);
    v16h b2 = frag16(&Ws[buf][nnb + 32][kb],   &Ws[buf][nnb + 32][16 + kb]);
    v16h b3 = frag16(&Ws[buf][nnb + 48][kb],   &Ws[buf][nnb + 48][16 + kb]);

    acc[0][0] = __builtin_amdgcn_wmma_f32_16x16x32_f16(false, a0, false, b0, (short)0, acc[0][0], false, false);
    acc[0][1] = __builtin_amdgcn_wmma_f32_16x16x32_f16(false, a0, false, b1, (short)0, acc[0][1], false, false);
    acc[0][2] = __builtin_amdgcn_wmma_f32_16x16x32_f16(false, a0, false, b2, (short)0, acc[0][2], false, false);
    acc[0][3] = __builtin_amdgcn_wmma_f32_16x16x32_f16(false, a0, false, b3, (short)0, acc[0][3], false, false);
    acc[1][0] = __builtin_amdgcn_wmma_f32_16x16x32_f16(false, a1, false, b0, (short)0, acc[1][0], false, false);
    acc[1][1] = __builtin_amdgcn_wmma_f32_16x16x32_f16(false, a1, false, b1, (short)0, acc[1][1], false, false);
    acc[1][2] = __builtin_amdgcn_wmma_f32_16x16x32_f16(false, a1, false, b2, (short)0, acc[1][2], false, false);
    acc[1][3] = __builtin_amdgcn_wmma_f32_16x16x32_f16(false, a1, false, b3, (short)0, acc[1][3], false, false);
    buf ^= 1;
  }

  // Epilogue. C layout: element r -> M = r + (lane>=16 ? 8:0), N = lane&15.
  const int rbase = (lane >> 4) * 8;
  const int col   = lane & 15;
#pragma unroll
  for (int i = 0; i < 2; i++) {
#pragma unroll
    for (int j = 0; j < 4; j++) {
      const int nOut = n0 + wn + j * 16 + col;
      const float bvv = bias ? bias[nOut] : 0.0f;
#pragma unroll
      for (int r = 0; r < 8; r++) {
        const int mOut = m0 + wm + i * 16 + rbase + r;
        const size_t ci = (size_t)z * strideC + (size_t)mOut * N + nOut;
        float v = acc[i][j][r] + bvv;
        if (accFlag) v += Cf[ci];
        if (actFlag) v = gelu_exact(v);
        if (Cf) Cf[ci] = v;
        if (Ch) Ch[ci] = (_Float16)v;
      }
    }
  }
}

// ---------------------------------------------------------------------------
// [B, D, L] -> [B, L, D] tiled transpose; x_enc -> f16, x_mark_enc -> f32
// ---------------------------------------------------------------------------
__global__ __launch_bounds__(256)
void transpose_in(const float* __restrict__ xe, const float* __restrict__ xme,
                  _Float16* __restrict__ x16, float* __restrict__ xmt)
{
  __shared__ float t1[32][33];
  __shared__ float t2[32][33];
  const int b = blockIdx.z, d0 = blockIdx.y * 32, l0 = blockIdx.x * 32;
  const int tx = threadIdx.x, ty = threadIdx.y;
#pragma unroll
  for (int i = 0; i < 4; i++) {
    const int d = d0 + ty + i * 8;
    const size_t src = ((size_t)b * Dn + d) * Ln + l0 + tx;
    t1[ty + i * 8][tx] = xe[src];
    t2[ty + i * 8][tx] = xme[src];
  }
  __syncthreads();
#pragma unroll
  for (int i = 0; i < 4; i++) {
    const int l = l0 + ty + i * 8;
    const size_t dst = ((size_t)b * Ln + l) * Dn + d0 + tx;
    x16[dst] = (_Float16)t1[tx][ty + i * 8];
    xmt[dst] = t2[tx][ty + i * 8];
  }
}

__global__ void cvt_f32_f16(const float* __restrict__ s, _Float16* __restrict__ d, int n)
{
  const int i = blockIdx.x * 256 + threadIdx.x;
  if (i < n) d[i] = (_Float16)s[i];
}

// W_emb [o,c,3] -> three [o,c] f16 slices
__global__ void cvt_emb(const float* __restrict__ W, _Float16* __restrict__ dst)
{
  const int i = blockIdx.x * 256 + threadIdx.x;
  if (i < 3 * Dn * Dn) {
    const int kk = i % 3, oc = i / 3;
    dst[kk * Dn * Dn + oc] = (_Float16)W[i];
  }
}

// mean_value[b,l] = (1/D) * sum_t G[b, t, (t-l) mod L]
__global__ __launch_bounds__(256)
void meanvalue_kernel(const float* __restrict__ G, float* __restrict__ mv)
{
  __shared__ float red[256];
  const int l = blockIdx.x, b = blockIdx.y, t = threadIdx.x;
  const float* Gb = G + (size_t)b * Ln * Ln;
  float s = Gb[(size_t)t * Ln + ((t - l) & (Ln - 1))]
          + Gb[(size_t)(t + 256) * Ln + ((t + 256 - l) & (Ln - 1))];
  red[t] = s; __syncthreads();
  for (int o = 128; o > 0; o >>= 1) { if (t < o) red[t] += red[t + o]; __syncthreads(); }
  if (t == 0) mv[b * Ln + l] = red[0] * (1.0f / 128.0f);
}

// top-18 lags of mean over b of mean_value (desc, lowest-index tie-break)
__global__ __launch_bounds__(512)
void topk_kernel(const float* __restrict__ mv, int* __restrict__ idx)
{
  __shared__ float val[Ln];
  __shared__ float rv[Ln];
  __shared__ int   ri[Ln];
  const int l = threadIdx.x;
  float s = 0.f;
  for (int b = 0; b < Bn; b++) s += mv[b * Ln + l];
  val[l] = s; __syncthreads();
  for (int k = 0; k < TOPK; k++) {
    rv[l] = val[l]; ri[l] = l; __syncthreads();
    for (int o = 256; o > 0; o >>= 1) {
      if (l < o) { if (rv[l + o] > rv[l]) { rv[l] = rv[l + o]; ri[l] = ri[l + o]; } }
      __syncthreads();
    }
    if (l == 0) { idx[k] = ri[0]; val[ri[0]] = -3.4e38f; }
    __syncthreads();
  }
}

__global__ void softmax_kernel(const float* __restrict__ mv, const int* __restrict__ idx,
                               float* __restrict__ w)
{
  const int b = blockIdx.x;
  if (threadIdx.x == 0) {
    float vals[TOPK]; float mx = -3.4e38f;
    for (int k = 0; k < TOPK; k++) { vals[k] = mv[b * Ln + idx[k]]; mx = fmaxf(mx, vals[k]); }
    float s = 0.f;
    for (int k = 0; k < TOPK; k++) { vals[k] = expf(vals[k] - mx); s += vals[k]; }
    const float inv = 1.0f / s;
    for (int k = 0; k < TOPK; k++) w[b * TOPK + k] = vals[k] * inv;
  }
}

// agg[b,l,d] = sum_k w[b,k] * v[b, (l+idx[k]) mod L, d]
__global__ __launch_bounds__(128)
void agg_kernel(const _Float16* __restrict__ v16, const float* __restrict__ w,
                const int* __restrict__ idx, _Float16* __restrict__ agg)
{
  __shared__ int   sidx[TOPK];
  __shared__ float sw[TOPK];
  const int l = blockIdx.x, b = blockIdx.y, d = threadIdx.x;
  if (d < TOPK) { sidx[d] = idx[d]; sw[d] = w[b * TOPK + d]; }
  __syncthreads();
  float acc = 0.f;
  for (int k = 0; k < TOPK; k++) {
    const int ls = (l + sidx[k]) & (Ln - 1);
    acc += sw[k] * (float)v16[((size_t)(b * Ln + ls)) * Dn + d];
  }
  agg[((size_t)(b * Ln + l)) * Dn + d] = (_Float16)acc;
}

// s = (p+q) - moving_avg_25(p+q)   (replicate pad == clamp); outputs f32 + f16
__global__ __launch_bounds__(256)
void add_decomp(const float* __restrict__ p, const float* __restrict__ q,
                float* __restrict__ sf, _Float16* __restrict__ sh)
{
  __shared__ float tile[16][Ln + 1];
  const int b = blockIdx.y, d0 = blockIdx.x * 16;
  const int tx = threadIdx.x & 15, ty = threadIdx.x >> 4;
  for (int l = ty; l < Ln; l += 16) {
    const size_t i = ((size_t)b * Ln + l) * Dn + d0 + tx;
    tile[tx][l] = p[i] + q[i];
  }
  __syncthreads();
  for (int l = ty; l < Ln; l += 16) {
    float sum = 0.f;
#pragma unroll
    for (int j = -12; j <= 12; j++) {
      int lj = l + j; lj = lj < 0 ? 0 : (lj > Ln - 1 ? Ln - 1 : lj);
      sum += tile[tx][lj];
    }
    const float o = tile[tx][l] - sum * (1.0f / 25.0f);
    const size_t i = ((size_t)b * Ln + l) * Dn + d0 + tx;
    sf[i] = o; sh[i] = (_Float16)o;
  }
}

__global__ __launch_bounds__(128)
void layernorm_kernel(const float* __restrict__ h, const float* __restrict__ gamma,
                      const float* __restrict__ beta, float* __restrict__ hn)
{
  __shared__ float red[Dn];
  const int r = blockIdx.x, d = threadIdx.x;
  const float v = h[(size_t)r * Dn + d];
  red[d] = v; __syncthreads();
  for (int o = 64; o > 0; o >>= 1) { if (d < o) red[d] += red[d + o]; __syncthreads(); }
  const float mu = red[0] * (1.0f / Dn); __syncthreads();
  const float df = v - mu;
  red[d] = df * df; __syncthreads();
  for (int o = 64; o > 0; o >>= 1) { if (d < o) red[d] += red[d + o]; __syncthreads(); }
  const float var = red[0] * (1.0f / Dn);
  hn[(size_t)r * Dn + d] = df * rsqrtf(var + 1e-5f) * gamma[d] + beta[d];
}

__global__ __launch_bounds__(128)
void colmean_kernel(const float* __restrict__ hn, float* __restrict__ cm)
{
  const int b = blockIdx.x, d = threadIdx.x;
  float s = 0.f;
  for (int l = 0; l < Ln; l++) s += hn[((size_t)b * Ln + l) * Dn + d];
  cm[b * Dn + d] = s * (1.0f / Ln);
}

__global__ __launch_bounds__(256)
void final_kernel(const float* __restrict__ hn, const float* __restrict__ cm,
                  const float* __restrict__ xmt, const float* __restrict__ pw,
                  const float* __restrict__ pb, float* __restrict__ out)
{
  __shared__ float red[256];
  const int b = blockIdx.x, t = threadIdx.x;
  const float* hb = hn  + (size_t)b * Ln * Dn;
  const float* xb = xmt + (size_t)b * Ln * Dn;
  const float* cb = cm  + (size_t)b * Dn;
  float acc = 0.f;
  for (int i = t; i < Ln * Dn; i += 256) {
    const float v = hb[i] - cb[i & (Dn - 1)];
    acc += gelu_exact(v) * xb[i] * pw[i];
  }
  red[t] = acc; __syncthreads();
  for (int o = 128; o > 0; o >>= 1) { if (t < o) red[t] += red[t + o]; __syncthreads(); }
  if (t == 0) out[b] = red[0] + pb[0];
}

// ---------------------------------------------------------------------------
extern "C" void kernel_launch(void* const* d_in, const int* in_sizes, int n_in,
                              void* d_out, int out_size, void* d_ws, size_t ws_size,
                              hipStream_t stream)
{
  (void)in_sizes; (void)n_in; (void)out_size; (void)ws_size;

  const float* x_enc  = (const float*)d_in[0];
  const float* xm_enc = (const float*)d_in[1];
  const float* W_emb  = (const float*)d_in[4];
  const float* Wq = (const float*)d_in[5];  const float* bq = (const float*)d_in[6];
  const float* Wk = (const float*)d_in[7];  const float* bk = (const float*)d_in[8];
  const float* Wv = (const float*)d_in[9];  const float* bv = (const float*)d_in[10];
  const float* Wo = (const float*)d_in[11]; const float* bo = (const float*)d_in[12];
  const float* W1 = (const float*)d_in[13]; const float* b1 = (const float*)d_in[14];
  const float* W2 = (const float*)d_in[15]; const float* b2 = (const float*)d_in[16];
  const float* gamma = (const float*)d_in[17];
  const float* beta  = (const float*)d_in[18];
  const float* projW = (const float*)d_in[19];
  const float* projb = (const float*)d_in[20];
  float* out = (float*)d_out;

  char* base = (char*)d_ws;
  size_t off = 0;
  auto alloc = [&](size_t bytes) -> char* {
    char* p = base + off;
    off += (bytes + 255) & ~(size_t)255;
    return p;
  };
  const int    NT  = Bn * Ln;            // 65536 rows
  const size_t BLD = (size_t)NT * Dn;    // 8,388,608 elems

  _Float16* x16   = (_Float16*)alloc(BLD * 2);
  float*    xmt   = (float*)   alloc(BLD * 4);
  float*    hbuf  = (float*)   alloc(BLD * 4);
  _Float16* h16   = (_Float16*)alloc(BLD * 2);
  _Float16* q16   = (_Float16*)alloc(BLD * 2);
  _Float16* k16   = (_Float16*)alloc(BLD * 2);
  _Float16* vv16  = (_Float16*)alloc(BLD * 2);
  // 134 MB region: G during autocorr, later y1(f16)+y2(f32)+hn(f32)
  char*     reg   = alloc((size_t)Bn * Ln * Ln * 4);
  float*    G     = (float*)reg;
  _Float16* y116  = (_Float16*)reg;
  float*    y2    = (float*)(reg + (size_t)NT * DFFn * 2);
  float*    hn    = (float*)(reg + (size_t)NT * DFFn * 2 + BLD * 4);
  float*    mv    = (float*)alloc((size_t)Bn * Ln * 4);
  int*      idx   = (int*)  alloc(256);
  float*    wsm   = (float*)alloc((size_t)Bn * TOPK * 4);
  _Float16* agg16 = (_Float16*)alloc(BLD * 2);
  float*    abuf  = (float*)alloc(BLD * 4);
  float*    sbuf  = (float*)alloc(BLD * 4);
  _Float16* s16   = (_Float16*)alloc(BLD * 2);
  float*    cmean = (float*)alloc((size_t)Bn * Dn * 4);
  _Float16* We16  = (_Float16*)alloc((size_t)3 * Dn * Dn * 2);
  _Float16* Wq16  = (_Float16*)alloc((size_t)ELn * Dn * Dn * 2);
  _Float16* Wk16  = (_Float16*)alloc((size_t)ELn * Dn * Dn * 2);
  _Float16* Wv16  = (_Float16*)alloc((size_t)ELn * Dn * Dn * 2);
  _Float16* Wo16  = (_Float16*)alloc((size_t)ELn * Dn * Dn * 2);
  _Float16* W116  = (_Float16*)alloc((size_t)ELn * DFFn * Dn * 2);
  _Float16* W216  = (_Float16*)alloc((size_t)ELn * Dn * DFFn * 2);

  // weight conversion to f16
  const int nDD = ELn * Dn * Dn, nFD = ELn * DFFn * Dn;
  cvt_f32_f16<<<(nDD + 255) / 256, 256, 0, stream>>>(Wq, Wq16, nDD);
  cvt_f32_f16<<<(nDD + 255) / 256, 256, 0, stream>>>(Wk, Wk16, nDD);
  cvt_f32_f16<<<(nDD + 255) / 256, 256, 0, stream>>>(Wv, Wv16, nDD);
  cvt_f32_f16<<<(nDD + 255) / 256, 256, 0, stream>>>(Wo, Wo16, nDD);
  cvt_f32_f16<<<(nFD + 255) / 256, 256, 0, stream>>>(W1, W116, nFD);
  cvt_f32_f16<<<(nFD + 255) / 256, 256, 0, stream>>>(W2, W216, nFD);
  cvt_emb<<<(3 * Dn * Dn + 255) / 256, 256, 0, stream>>>(W_emb, We16);

  transpose_in<<<dim3(Ln / 32, Dn / 32, Bn), dim3(32, 8), 0, stream>>>(
      x_enc, xm_enc, x16, xmt);

  auto gemm = [&](const _Float16* A, int lda, long long sA,
                  const _Float16* Wp, int ldw, long long sW,
                  const float* bias, float* Cf, _Float16* Ch, long long sC,
                  int M, int N, int K, int shift, int acc, int act, int nb) {
    gemm_wmma<<<dim3(N / 128, M / 128, nb), 256, 0, stream>>>(
        A, lda, sA, Wp, ldw, sW, bias, Cf, Ch, sC, M, N, K, Ln, shift, acc, act);
  };

  // token embedding: circular conv k=3 as 3 accumulating GEMM calls
  gemm(x16, Dn, 0, We16 + 0 * Dn * Dn, Dn, 0, nullptr, hbuf, nullptr, 0,
       NT, Dn, Dn, -1, 0, 0, 1);
  gemm(x16, Dn, 0, We16 + 1 * Dn * Dn, Dn, 0, nullptr, hbuf, nullptr, 0,
       NT, Dn, Dn,  0, 1, 0, 1);
  gemm(x16, Dn, 0, We16 + 2 * Dn * Dn, Dn, 0, nullptr, hbuf, h16, 0,
       NT, Dn, Dn, +1, 1, 0, 1);

  for (int l = 0; l < ELn; l++) {
    // Q, K, V projections (f16 outputs)
    gemm(h16, Dn, 0, Wq16 + l * Dn * Dn, Dn, 0, bq + l * Dn, nullptr, q16, 0,
         NT, Dn, Dn, 0, 0, 0, 1);
    gemm(h16, Dn, 0, Wk16 + l * Dn * Dn, Dn, 0, bk + l * Dn, nullptr, k16, 0,
         NT, Dn, Dn, 0, 0, 0, 1);
    gemm(h16, Dn, 0, Wv16 + l * Dn * Dn, Dn, 0, bv + l * Dn, nullptr, vv16, 0,
         NT, Dn, Dn, 0, 0, 0, 1);
    // autocorrelation via batched Gram GEMM G[b] = Q_b K_b^T  (FFT-free)
    gemm(q16, Dn, (long long)Ln * Dn, k16, Dn, (long long)Ln * Dn, nullptr,
         G, nullptr, (long long)Ln * Ln, Ln, Ln, Dn, 0, 0, 0, Bn);
    meanvalue_kernel<<<dim3(Ln, Bn), 256, 0, stream>>>(G, mv);
    topk_kernel<<<1, 512, 0, stream>>>(mv, idx);
    softmax_kernel<<<Bn, 32, 0, stream>>>(mv, idx, wsm);
    agg_kernel<<<dim3(Ln, Bn), Dn, 0, stream>>>(vv16, wsm, idx, agg16);
    // output projection
    gemm(agg16, Dn, 0, Wo16 + l * Dn * Dn, Dn, 0, bo + l * Dn, abuf, nullptr, 0,
         NT, Dn, Dn, 0, 0, 0, 1);
    // decomp1: s = decomp(h + a)
    add_decomp<<<dim3(Dn / 16, Bn), 256, 0, stream>>>(hbuf, abuf, sbuf, s16);
    // FFN: y1 = gelu(s W1^T + b1) [f16], y2 = y1 W2^T + b2
    gemm(s16, Dn, 0, W116 + l * DFFn * Dn, Dn, 0, b1 + l * DFFn, nullptr, y116, 0,
         NT, DFFn, Dn, 0, 0, 1, 1);
    gemm(y116, DFFn, 0, W216 + l * Dn * DFFn, DFFn, 0, b2 + l * Dn, y2, nullptr, 0,
         NT, Dn, DFFn, 0, 0, 0, 1);
    // decomp2: h = decomp(s + y2)
    add_decomp<<<dim3(Dn / 16, Bn), 256, 0, stream>>>(sbuf, y2, hbuf, h16);
  }

  layernorm_kernel<<<NT, Dn, 0, stream>>>(hbuf, gamma, beta, hn);
  colmean_kernel<<<Bn, Dn, 0, stream>>>(hn, cmean);
  final_kernel<<<Bn, 256, 0, stream>>>(hn, cmean, xmt, projW, projb, out);
}
